// Deformable_conv_45414984188205
// MI455X (gfx1250) — matrix operand order repaired and checked by reading, not compiled, not run on hardware
//
#include <hip/hip_runtime.h>

// ---------------------------------------------------------------------------
// Deformable conv v2 on gfx1250 (MI455X), bf16 WMMA pipeline.
//   B=8, C=256, G=4, Cg=64, H=W=64, stride=1, pad=1, 3x3 kernels.
// ---------------------------------------------------------------------------

typedef __attribute__((ext_vector_type(16))) __bf16 v16bf;
typedef __attribute__((ext_vector_type(8)))  float  v8f;

#define BDIM 8
#define CDIM 256
#define GDIM 4
#define CG   64
#define HDIM 64
#define WDIM 64
#define HW   (HDIM*WDIM)          // 4096
#define NOFF 72
#define NOM  108                  // 72 offset + 36 mask channels
#define NOMP 112                  // padded to 7 N-tiles of 16
#define KSTEPS_OM 72              // K = 9*256 = 2304, /32
#define KSTEPS_CV 18              // K = 9*64  = 576,  /32

// workspace layout (bytes), all offsets 256B aligned
#define WS_XBF   0ull                                            // 16,777,216
#define WS_WOFM  (WS_XBF   + (size_t)BDIM*GDIM*HW*CG*2)          // +  516,096
#define WS_WCONV (WS_WOFM  + (size_t)KSTEPS_OM*7*32*16*2)        // +  294,912
#define WS_OFFM  (WS_WCONV + (size_t)GDIM*4*KSTEPS_CV*32*16*2)   // + 14,680,064

union FragB { v16bf v; uint4 q[2]; unsigned short u[16]; };
union U16x8 { uint4 q; unsigned short u[8]; };

__device__ __forceinline__ float bf2f(unsigned short h) {
    return __uint_as_float(((unsigned)h) << 16);
}
__device__ __forceinline__ unsigned short f2bf(float f) {
    unsigned u = __float_as_uint(f);
    u += 0x7fffu + ((u >> 16) & 1u);           // round-to-nearest-even
    return (unsigned short)(u >> 16);
}

// --------------------------- prep: x -> bf16 (B,G,HW,Cg) -------------------
__global__ __launch_bounds__(256) void k_prep_x(const float* __restrict__ x,
                                                unsigned short* __restrict__ xbf) {
    int i = blockIdx.x * 256 + threadIdx.x;    // total = 8*4*4096*64 = 8388608
    int cg   = i & 63;
    int rest = i >> 6;
    int pix  = rest & (HW - 1);
    rest >>= 12;
    int g = rest & 3;
    int b = rest >> 2;
    float v = x[((size_t)b * CDIM + g * CG + cg) * HW + pix];
    xbf[i] = f2bf(v);
}

// ------------- prep: offset+mask weights -> WMMA B-operand fragments -------
// layout: ((ks*7 + nt)*32 + lane)*16 + j ; B frag: lane holds K = hi*16 + j
__global__ __launch_bounds__(256) void k_prep_wofm(const float* __restrict__ w_off,
                                                   const float* __restrict__ w_msk,
                                                   unsigned short* __restrict__ wf) {
    int i = blockIdx.x * 256 + threadIdx.x;    // total = 72*7*512 = 258048
    int j    = i & 15;
    int lane = (i >> 4) & 31;
    int rest = i >> 9;
    int nt   = rest % 7;
    int ks   = rest / 7;
    int hi   = lane >> 4;
    int n    = nt * 16 + (lane & 15);
    int K    = ks * 32 + hi * 16 + j;          // K-order: tap-major, 256 ch/tap
    int tap  = K >> 8;
    int c    = K & 255;
    int ty = tap / 3, tx = tap % 3;
    float v = 0.f;
    if (n < NOFF)     v = w_off[(((size_t)n * CDIM + c) * 3 + ty) * 3 + tx];
    else if (n < NOM) v = w_msk[(((size_t)(n - NOFF) * CDIM + c) * 3 + ty) * 3 + tx];
    wf[i] = f2bf(v);
}

// ------------- prep: grouped conv weights -> WMMA A-operand fragments ------
// layout: (((g*4+ot)*18 + ks)*32 + lane)*16 + j
// A frag (16x32 bf16): lane holds K = hi*16 + j
__global__ __launch_bounds__(256) void k_prep_wconv(const float* __restrict__ w_conv,
                                                    unsigned short* __restrict__ wf) {
    int i = blockIdx.x * 256 + threadIdx.x;    // total = 4*4*18*512 = 147456
    int j    = i & 15;
    int lane = (i >> 4) & 31;
    int rest = i >> 9;
    int ks   = rest % 18;
    int got  = rest / 18;
    int ot   = got & 3;
    int g    = got >> 2;
    int hi   = lane >> 4;
    int m    = ot * 16 + (lane & 15);          // out-channel within group
    int klocal = j + hi * 16;
    int K    = ks * 32 + klocal;               // K-order: tap-major, 64 ch/tap
    int tap  = K >> 6;
    int c    = K & 63;
    int ty = tap / 3, tx = tap % 3;
    float v = w_conv[(((size_t)(g * CG + m) * CG + c) * 3 + ty) * 3 + tx];
    wf[i] = f2bf(v);
}

// ----------------- stage 1: offset/mask conv as implicit GEMM --------------
// grid = B*32 blocks, 256 thr (8 waves). Wave owns 16 pixels x 7 N-tiles.
__global__ __launch_bounds__(256) void k_offmask(const unsigned short* __restrict__ xbf,
                                                 const unsigned short* __restrict__ wofm,
                                                 const float* __restrict__ b_off,
                                                 const float* __restrict__ b_msk,
                                                 float* __restrict__ offm) {
    const int lane = threadIdx.x & 31;
    const int wave = threadIdx.x >> 5;
    const int col  = lane & 15;
    const int hi   = lane >> 4;
    const int b      = blockIdx.x >> 5;
    const int pixblk = blockIdx.x & 31;
    const int wbase  = pixblk * 128 + wave * 16;
    const int pix = wbase + col;               // 16 contiguous px -> same row
    const int h = pix >> 6, w = pix & 63;
    const unsigned short* xb = xbf + (size_t)b * GDIM * HW * CG;

    v8f acc[7] = {};

    for (int ks = 0; ks < KSTEPS_OM; ++ks) {
        int tap = ks >> 3;                     // 8 chunks of 32 channels per tap
        int cc  = ks & 7;
        int g   = cc >> 1;
        int cgb = (cc & 1) * 32;
        int hy = h + tap / 3 - 1;
        int wx = w + tap % 3 - 1;
        bool inb = ((unsigned)hy < (unsigned)HDIM) && ((unsigned)wx < (unsigned)WDIM);
        int hyc = min(max(hy, 0), HDIM - 1);
        int wxc = min(max(wx, 0), WDIM - 1);
        const unsigned short* p =
            xb + (((size_t)g * HW + hyc * WDIM + wxc) * CG + cgb + hi * 16);
        FragB a;
        a.q[0] = *(const uint4*)p;             // K 0..7 / 16..23
        a.q[1] = *(const uint4*)(p + 8);       // K 8..15 / 24..31
        if (!inb) { a.q[0] = make_uint4(0,0,0,0); a.q[1] = make_uint4(0,0,0,0); }

        #pragma unroll
        for (int nt = 0; nt < 7; ++nt) {
            FragB bf_;
            const uint4* wq =
                (const uint4*)(wofm + ((size_t)(ks * 7 + nt) * 32 + lane) * 16);
            bf_.q[0] = wq[0];
            bf_.q[1] = wq[1];
            acc[nt] = __builtin_amdgcn_wmma_f32_16x16x32_bf16(
                false, a.v, false, bf_.v, (short)0, acc[nt], false, false);
        }
    }

    // epilogue: D[m=pixel, n=channel]; lane holds n = nt*16+col, m = r + 8*hi
    float bias[7];
    #pragma unroll
    for (int nt = 0; nt < 7; ++nt) {
        int n = nt * 16 + col;
        bias[nt] = (n < NOFF) ? b_off[n] : ((n < NOM) ? b_msk[n - NOFF] : 0.f);
    }
    float* orow = offm + (size_t)b * HW * NOMP;
    #pragma unroll
    for (int nt = 0; nt < 7; ++nt) {
        int n = nt * 16 + col;
        if (n < NOM) {
            #pragma unroll
            for (int r = 0; r < 8; ++r) {
                int mpix = wbase + r + 8 * hi;
                orow[(size_t)mpix * NOMP + n] = acc[nt][r] + bias[nt];
            }
        }
    }
}

// ------------- stage 2: fused bilinear sampling + grouped GEMM -------------
// grid = B*G*32 blocks, 256 thr. Wave owns 16 pixels x 4 o-tiles (full Cg=64).
__global__ __launch_bounds__(256) void k_deform(const unsigned short* __restrict__ xbf,
                                                const unsigned short* __restrict__ wcf,
                                                const float* __restrict__ offm,
                                                float* __restrict__ out) {
    const int lane = threadIdx.x & 31;
    const int wave = threadIdx.x >> 5;
    const int col  = lane & 15;
    const int hi   = lane >> 4;
    const int blk    = blockIdx.x;
    const int pixblk = blk & 31;
    const int g      = (blk >> 5) & 3;
    const int b      = blk >> 7;
    const int wbase  = pixblk * 128 + wave * 16;
    const int pix = wbase + col;
    const int h = pix >> 6, w = pix & 63;
    const unsigned short* xb = xbf + ((size_t)(b * GDIM + g)) * HW * CG;
    const float* om = offm + ((size_t)b * HW + pix) * NOMP;

    v8f acc[4] = {};

    for (int tap = 0; tap < 9; ++tap) {
        float dy  = om[(g * 9 + tap) * 2 + 0];
        float dx  = om[(g * 9 + tap) * 2 + 1];
        float mr  = om[NOFF + g * 9 + tap];
        float msk = 1.0f / (1.0f + __expf(-mr));   // sigmoid(mask)

        float py  = (float)(h - 1 + tap / 3) + dy;
        float pxf = (float)(w - 1 + tap % 3) + dx;
        float y0f = floorf(py), x0f = floorf(pxf);
        float ly = py - y0f, lx = pxf - x0f;
        int y0 = (int)y0f, x0 = (int)x0f;
        int y1 = y0 + 1,  x1 = x0 + 1;
        bool vy0 = (unsigned)y0 < (unsigned)HDIM;
        bool vy1 = (unsigned)y1 < (unsigned)HDIM;
        bool vx0 = (unsigned)x0 < (unsigned)WDIM;
        bool vx1 = (unsigned)x1 < (unsigned)WDIM;
        // mask folded into the 4 corner weights: zero extra per-channel cost
        float w00 = (vy0 && vx0) ? (1.f - ly) * (1.f - lx) * msk : 0.f;
        float w01 = (vy0 && vx1) ? (1.f - ly) * lx * msk         : 0.f;
        float w10 = (vy1 && vx0) ? ly * (1.f - lx) * msk         : 0.f;
        float w11 = (vy1 && vx1) ? ly * lx * msk                 : 0.f;
        int y0c = min(max(y0, 0), HDIM - 1), y1c = min(max(y1, 0), HDIM - 1);
        int x0c = min(max(x0, 0), WDIM - 1), x1c = min(max(x1, 0), WDIM - 1);
        const unsigned short* p00 = xb + (size_t)(y0c * WDIM + x0c) * CG;
        const unsigned short* p01 = xb + (size_t)(y0c * WDIM + x1c) * CG;
        const unsigned short* p10 = xb + (size_t)(y1c * WDIM + x0c) * CG;
        const unsigned short* p11 = xb + (size_t)(y1c * WDIM + x1c) * CG;

        #pragma unroll
        for (int st = 0; st < 2; ++st) {           // 2 x 32 channels per tap
            const int cb = st * 32 + hi * 16;      // B frag: lane K = hi*16 + j
            U16x8 s00a, s00b, s01a, s01b, s10a, s10b, s11a, s11b;
            s00a.q = *(const uint4*)(p00 + cb); s00b.q = *(const uint4*)(p00 + cb + 8);
            s01a.q = *(const uint4*)(p01 + cb); s01b.q = *(const uint4*)(p01 + cb + 8);
            s10a.q = *(const uint4*)(p10 + cb); s10b.q = *(const uint4*)(p10 + cb + 8);
            s11a.q = *(const uint4*)(p11 + cb); s11b.q = *(const uint4*)(p11 + cb + 8);

            FragB bf_;
            #pragma unroll
            for (int j = 0; j < 8; ++j) {
                float f0 = w00 * bf2f(s00a.u[j]) + w01 * bf2f(s01a.u[j])
                         + w10 * bf2f(s10a.u[j]) + w11 * bf2f(s11a.u[j]);
                float f1 = w00 * bf2f(s00b.u[j]) + w01 * bf2f(s01b.u[j])
                         + w10 * bf2f(s10b.u[j]) + w11 * bf2f(s11b.u[j]);
                bf_.u[j]     = f2bf(f0);
                bf_.u[8 + j] = f2bf(f1);
            }

            const int ks = tap * 2 + st;
            #pragma unroll
            for (int ot = 0; ot < 4; ++ot) {       // B frag reused 4x
                FragB af;
                const uint4* aq = (const uint4*)(wcf +
                    (((size_t)((g * 4 + ot) * KSTEPS_CV + ks)) * 32 + lane) * 16);
                af.q[0] = aq[0];
                af.q[1] = aq[1];
                acc[ot] = __builtin_amdgcn_wmma_f32_16x16x32_bf16(
                    false, af.v, false, bf_.v, (short)0, acc[ot], false, false);
            }
        }
    }

    // epilogue: D[m=out-ch, n=pixel]; lane holds n=col, rows m = r + 8*hi
    float* op = out + ((size_t)b * CDIM + g * CG) * HW;
    #pragma unroll
    for (int ot = 0; ot < 4; ++ot) {
        #pragma unroll
        for (int r = 0; r < 8; ++r) {
            op[(size_t)(ot * 16 + r + 8 * hi) * HW + pix] = acc[ot][r];
        }
    }
}

// ---------------------------------------------------------------------------
extern "C" void kernel_launch(void* const* d_in, const int* in_sizes, int n_in,
                              void* d_out, int out_size, void* d_ws, size_t ws_size,
                              hipStream_t stream) {
    (void)in_sizes; (void)n_in; (void)out_size; (void)ws_size;
    const float* x      = (const float*)d_in[0];
    const float* w_off  = (const float*)d_in[1];
    const float* b_off  = (const float*)d_in[2];
    const float* w_msk  = (const float*)d_in[3];
    const float* b_msk  = (const float*)d_in[4];
    const float* w_conv = (const float*)d_in[5];
    float* out = (float*)d_out;

    char* ws = (char*)d_ws;
    unsigned short* xbf  = (unsigned short*)(ws + WS_XBF);
    unsigned short* wofm = (unsigned short*)(ws + WS_WOFM);
    unsigned short* wcf  = (unsigned short*)(ws + WS_WCONV);
    float*          offm = (float*)(ws + WS_OFFM);

    k_prep_x    <<<32768, 256, 0, stream>>>(x, xbf);
    k_prep_wofm <<<1008,  256, 0, stream>>>(w_off, w_msk, wofm);
    k_prep_wconv<<<576,   256, 0, stream>>>(w_conv, wcf);
    k_offmask   <<<BDIM * 32,        256, 0, stream>>>(xbf, wofm, b_off, b_msk, offm);
    k_deform    <<<BDIM * GDIM * 32, 256, 0, stream>>>(xbf, wcf, offm, out);
}
